// MultiHeadAttention_28278064677417
// MI455X (gfx1250) — matrix operand order repaired and checked by reading, not compiled, not run on hardware
//
#include <hip/hip_runtime.h>
#include <hip/hip_bf16.h>

// CDNA5 / gfx1250, wave32. All matmuls via v_wmma_f32_16x16x32_f16.
// Attention stages K/V tiles into LDS with async global->LDS loads
// (ASYNCcnt / s_wait_asynccnt) when the toolchain provides the builtin.
typedef __attribute__((ext_vector_type(16))) _Float16 v16h;
typedef __attribute__((ext_vector_type(8)))  float    v8f;
typedef __attribute__((__vector_size__(16))) int      vi4;   // int4 for async LDS ops

#define D_MODEL 1024
#define N_HEADS 16
#define D_HEAD  64
#define BATCH   2
#define SEQ     2048
#define NEG_BIG (-1.0e30f)

#define __AS1 __attribute__((address_space(1)))
#define __AS3 __attribute__((address_space(3)))

#if __has_builtin(__builtin_amdgcn_global_load_async_to_lds_b128)
#define HAVE_ASYNC_LDS 1
#else
#define HAVE_ASYNC_LDS 0
#endif

static __device__ __forceinline__ v8f wmma_f16f32(v16h a, v16h b, v8f c) {
  // D = A(16x32 f16) * B(32x16 f16) + C(16x16 f32)
  return __builtin_amdgcn_wmma_f32_16x16x32_f16(false, a, false, b, (short)0, c,
                                                false, false);
}

static __device__ __forceinline__ float half_shfl_max(float v) {
  // reduce across the 16-lane half-wave (xor masks never touch bit4)
  v = fmaxf(v, __shfl_xor(v, 1, 32));
  v = fmaxf(v, __shfl_xor(v, 2, 32));
  v = fmaxf(v, __shfl_xor(v, 4, 32));
  v = fmaxf(v, __shfl_xor(v, 8, 32));
  return v;
}
static __device__ __forceinline__ float half_shfl_sum(float v) {
  v += __shfl_xor(v, 1, 32);
  v += __shfl_xor(v, 2, 32);
  v += __shfl_xor(v, 4, 32);
  v += __shfl_xor(v, 8, 32);
  return v;
}

// ---------------------------------------------------------------------------
// Q/K/V projection: Out_f16[b][h][l][d] = (X @ W^T + bias), cast to f16.
// One wave -> 16 rows x 64 cols (one head slice), K loop in chunks of 32.
// ---------------------------------------------------------------------------
__global__ __launch_bounds__(256) void qkv_proj_kernel(
    const float* __restrict__ X,     // [BATCH*SEQ, D_MODEL] fp32
    const float* __restrict__ W,     // [D_MODEL, D_MODEL] fp32 (row n = W[n,:])
    const float* __restrict__ bias,  // [D_MODEL]
    _Float16* __restrict__ Out)      // [BATCH, N_HEADS, SEQ, D_HEAD] f16
{
  const int lane = threadIdx.x & 31;
  const int wave = blockIdx.x * (blockDim.x >> 5) + (threadIdx.x >> 5);
  const int MT = (BATCH * SEQ) / 16;      // 256 row tiles
  const int wm = wave % MT;
  const int wn = wave / MT;               // 0..15 -> 64-wide column strip
  const int m0 = wm * 16;
  const int c0 = wn * 64;

  const int m  = lane & 15;
  const int hi = lane >> 4;
  const int n  = lane & 15;
  const int ka = hi << 4;   // A-layout K offset (0 or 16)
  const int kb = hi << 4;   // B-layout K offset (0 or 16)

  v8f acc[4];
  #pragma unroll
  for (int s = 0; s < 4; ++s) {
    float bvl = bias[c0 + 16 * s + n];
    #pragma unroll
    for (int r = 0; r < 8; ++r) acc[s][r] = bvl;
  }

  const float* arow = X + (size_t)(m0 + m) * D_MODEL + ka;
  #pragma unroll 1
  for (int k0 = 0; k0 < D_MODEL; k0 += 32) {
    v16h a;
    #pragma unroll
    for (int i = 0; i < 8; ++i) {
      a[i]     = (_Float16)arow[k0 + i];        // K = k0 + ka + i
      a[i + 8] = (_Float16)arow[k0 + 8 + i];    // K = k0 + ka + 8 + i
    }
    #pragma unroll
    for (int s = 0; s < 4; ++s) {
      const float* brow = W + (size_t)(c0 + 16 * s + n) * D_MODEL + k0 + kb;
      v16h bm;
      #pragma unroll
      for (int i = 0; i < 16; ++i) bm[i] = (_Float16)brow[i];  // K = k0+kb+i
      acc[s] = wmma_f16f32(a, bm, acc[s]);
    }
  }

  const int h = c0 >> 6;  // head index (c0 is 64-aligned)
  #pragma unroll
  for (int s = 0; s < 4; ++s) {
    const int d = 16 * s + n;
    #pragma unroll
    for (int r = 0; r < 8; ++r) {
      int t = m0 + hi * 8 + r;
      int b = t >> 11;            // / SEQ
      int l = t & (SEQ - 1);
      Out[(((size_t)b * N_HEADS + h) * SEQ + l) * D_HEAD + d] =
          (_Float16)acc[s][r];
    }
  }
}

// ---------------------------------------------------------------------------
// Flash attention. One block (8 waves) handles 8 q-tiles of one (b,h).
// K/V 32-key tiles are cooperatively staged into LDS, double-buffered with
// async global->LDS loads; each wave runs the online-softmax WMMA pipeline.
// ---------------------------------------------------------------------------
__global__ __launch_bounds__(256) void mha_attn_kernel(
    const _Float16* __restrict__ Qh, const _Float16* __restrict__ Kh,
    const _Float16* __restrict__ Vh, const unsigned char* __restrict__ kmask,
    _Float16* __restrict__ AttOut)   // [BATCH*SEQ, D_MODEL] token-major f16
{
  __shared__ _Float16 lds_k[2][32 * D_HEAD];  // 2 x 4 KB
  __shared__ _Float16 lds_v[2][32 * D_HEAD];  // 2 x 4 KB
  __shared__ _Float16 lds_p[8][16 * 36];      // per-wave 16x32 P tile

  const int tid  = threadIdx.x;
  const int lane = tid & 31;
  const int warp = tid >> 5;

  // 16 blocks per (b,h); 8 q-tiles per block (one per wave)
  const int bh     = blockIdx.x >> 4;
  const int qstrip = blockIdx.x & 15;
  const int b = bh >> 4;
  const int h = bh & 15;
  const int qt = qstrip * 8 + warp;

  const size_t headBase = ((size_t)b * N_HEADS + h) * SEQ * D_HEAD;
  const _Float16* Qbase = Qh + headBase + (size_t)qt * 16 * D_HEAD;
  const _Float16* Kbase = Kh + headBase;
  const _Float16* Vbase = Vh + headBase;
  const unsigned char* mrow = kmask + (size_t)b * SEQ;

  const int m  = lane & 15;
  const int hi = lane >> 4;
  const int n  = lane & 15;
  const int ka = hi << 4;   // A-layout K offset
  const int kb = hi << 4;   // B-layout K offset

  // Q tile, 16x64 f16 -> two A operands (d 0..31, d 32..63)
  v16h aq0, aq1;
  {
    const _Float16* qr = Qbase + m * D_HEAD + ka;
    #pragma unroll
    for (int i = 0; i < 8; ++i) {
      aq0[i]     = qr[i];
      aq0[i + 8] = qr[8 + i];
      aq1[i]     = qr[32 + i];
      aq1[i + 8] = qr[40 + i];
    }
  }

  v8f o0 = {}, o1 = {}, o2 = {}, o3 = {};
  float mstat[8], lstat[8];
  #pragma unroll
  for (int r = 0; r < 8; ++r) { mstat[r] = NEG_BIG; lstat[r] = 0.f; }

  _Float16* pbuf = &lds_p[warp][0];

  // Cooperative stage of one 32-key K+V tile: 256 threads x 16 B each buffer.
  auto stage = [&](int kt, int buf) {
    const _Float16* gk = Kbase + (size_t)kt * D_HEAD + tid * 8;
    const _Float16* gv = Vbase + (size_t)kt * D_HEAD + tid * 8;
    _Float16* lk = &lds_k[buf][tid * 8];
    _Float16* lv = &lds_v[buf][tid * 8];
#if HAVE_ASYNC_LDS
    __builtin_amdgcn_global_load_async_to_lds_b128(
        (__AS1 vi4*)gk, (__AS3 vi4*)lk, 0, 0);
    __builtin_amdgcn_global_load_async_to_lds_b128(
        (__AS1 vi4*)gv, (__AS3 vi4*)lv, 0, 0);
#else
    *(vi4*)lk = *(const vi4*)gk;
    *(vi4*)lv = *(const vi4*)gv;
#endif
  };

  stage(0, 0);

  #pragma unroll 1
  for (int kt = 0; kt < SEQ; kt += 32) {
    const int buf = (kt >> 5) & 1;
    if (kt + 32 < SEQ) {
      stage(kt + 32, buf ^ 1);   // prefill next buffer
#if HAVE_ASYNC_LDS
      // 4 async ops outstanding (2 current + 2 next); in-order completion:
      // cnt<=2 guarantees the current tile's loads landed in LDS.
#if __has_builtin(__builtin_amdgcn_s_wait_asynccnt)
      __builtin_amdgcn_s_wait_asynccnt(2);
#else
      asm volatile("s_wait_asynccnt 0x2" ::: "memory");
#endif
#endif
    } else {
#if HAVE_ASYNC_LDS
#if __has_builtin(__builtin_amdgcn_s_wait_asynccnt)
      __builtin_amdgcn_s_wait_asynccnt(0);
#else
      asm volatile("s_wait_asynccnt 0x0" ::: "memory");
#endif
#endif
    }
    __syncthreads();  // all waves' staging for this tile complete

    const _Float16* lk = &lds_k[buf][0];
    const _Float16* lv = &lds_v[buf][0];

    // --- K^T B-tiles from LDS: element(kk, n) = K[kt+n][d], contiguous d ---
    const _Float16* k0p = lk + n * D_HEAD + kb;
    const _Float16* k1p = k0p + 16 * D_HEAD;
    v16h bk00, bk01, bk10, bk11;
    #pragma unroll
    for (int i = 0; i < 16; ++i) {
      bk00[i] = k0p[i];        // keys kt+n,    d = kb + i
      bk01[i] = k0p[32 + i];   //               d = 32 + kb + i
      bk10[i] = k1p[i];        // keys kt+16+n
      bk11[i] = k1p[32 + i];
    }
    v8f z = {};
    v8f s0 = wmma_f16f32(aq0, bk00, z);
    s0 = wmma_f16f32(aq1, bk01, s0);
    v8f s1 = wmma_f16f32(aq0, bk10, z);
    s1 = wmma_f16f32(aq1, bk11, s1);

    const float sc = 0.125f;  // 1/sqrt(D_HEAD)
    const bool ok0 = mrow[kt + n] != 0;
    const bool ok1 = mrow[kt + 16 + n] != 0;

    // --- online softmax (lane holds 1 col x 8 rows per tile) ---
    float al[8];
    #pragma unroll
    for (int r = 0; r < 8; ++r) {
      float v0 = ok0 ? s0[r] * sc : NEG_BIG;
      float v1 = ok1 ? s1[r] * sc : NEG_BIG;
      float rowmax = half_shfl_max(fmaxf(v0, v1));
      float mnew = fmaxf(mstat[r], rowmax);
      al[r] = __expf(mstat[r] - mnew);
      float p0 = __expf(v0 - mnew);
      float p1 = __expf(v1 - mnew);
      lstat[r] = al[r] * lstat[r] + half_shfl_sum(p0 + p1);
      mstat[r] = mnew;
      int Mr = hi * 8 + r;
      pbuf[Mr * 36 + n]      = (_Float16)p0;
      pbuf[Mr * 36 + 16 + n] = (_Float16)p1;
    }
    #pragma unroll
    for (int r = 0; r < 8; ++r) {
      o0[r] *= al[r]; o1[r] *= al[r]; o2[r] *= al[r]; o3[r] *= al[r];
    }

    // same-wave LDS RAW on the P tile (per-wave region, DS ops in-order)
    asm volatile("s_wait_dscnt 0x0" ::: "memory");

    // --- P A-tile from LDS (C-layout -> A-layout transpose) ---
    v16h ap;
    {
      const _Float16* pr = pbuf + m * 36 + ka;
      #pragma unroll
      for (int i = 0; i < 8; ++i) { ap[i] = pr[i]; ap[i + 8] = pr[8 + i]; }
    }

    // --- V B-tiles from LDS: element(kk, n) = V[kt+kk][16*dt + n] ---
    const _Float16* vp = lv + kb * D_HEAD + n;
    v16h bv0, bv1, bv2, bv3;
    #pragma unroll
    for (int i = 0; i < 16; ++i) {
      const _Float16* vr = vp + i * D_HEAD;
      bv0[i] = vr[0];
      bv1[i] = vr[16];
      bv2[i] = vr[32];
      bv3[i] = vr[48];
    }
    o0 = wmma_f16f32(ap, bv0, o0);
    o1 = wmma_f16f32(ap, bv1, o1);
    o2 = wmma_f16f32(ap, bv2, o2);
    o3 = wmma_f16f32(ap, bv3, o3);

    __syncthreads();  // compute done before this buffer is refilled
  }

  // normalize and store token-major f16 (row = token, col = h*64 + d)
  #pragma unroll
  for (int r = 0; r < 8; ++r) {
    float inv = lstat[r] > 0.f ? 1.f / lstat[r] : 0.f;
    size_t row = ((size_t)b * SEQ + qt * 16 + hi * 8 + r) * D_MODEL
               + (size_t)h * D_HEAD;
    AttOut[row + n]      = (_Float16)(o0[r] * inv);
    AttOut[row + 16 + n] = (_Float16)(o1[r] * inv);
    AttOut[row + 32 + n] = (_Float16)(o2[r] * inv);
    AttOut[row + 48 + n] = (_Float16)(o3[r] * inv);
  }
}

// ---------------------------------------------------------------------------
// Output projection: out_fp32 = att_f16 @ Wo^T + bo
// ---------------------------------------------------------------------------
__global__ __launch_bounds__(256) void out_proj_kernel(
    const _Float16* __restrict__ A,  // [BATCH*SEQ, D_MODEL] f16 token-major
    const float* __restrict__ W,     // Wo [D_MODEL, D_MODEL] fp32
    const float* __restrict__ bias,  // bo
    float* __restrict__ Out)         // [BATCH*SEQ, D_MODEL] fp32
{
  const int lane = threadIdx.x & 31;
  const int wave = blockIdx.x * (blockDim.x >> 5) + (threadIdx.x >> 5);
  const int MT = (BATCH * SEQ) / 16;
  const int wm = wave % MT;
  const int wn = wave / MT;
  const int m0 = wm * 16;
  const int c0 = wn * 64;

  const int m  = lane & 15;
  const int hi = lane >> 4;
  const int n  = lane & 15;
  const int ka = hi << 4;
  const int kb = hi << 4;

  v8f acc[4];
  #pragma unroll
  for (int s = 0; s < 4; ++s) {
    float bvl = bias[c0 + 16 * s + n];
    #pragma unroll
    for (int r = 0; r < 8; ++r) acc[s][r] = bvl;
  }

  const _Float16* arow = A + (size_t)(m0 + m) * D_MODEL + ka;
  #pragma unroll 1
  for (int k0 = 0; k0 < D_MODEL; k0 += 32) {
    v16h a;
    #pragma unroll
    for (int i = 0; i < 8; ++i) {
      a[i]     = arow[k0 + i];
      a[i + 8] = arow[k0 + 8 + i];
    }
    #pragma unroll
    for (int s = 0; s < 4; ++s) {
      const float* brow = W + (size_t)(c0 + 16 * s + n) * D_MODEL + k0 + kb;
      v16h bm;
      #pragma unroll
      for (int i = 0; i < 16; ++i) bm[i] = (_Float16)brow[i];
      acc[s] = wmma_f16f32(a, bm, acc[s]);
    }
  }

  #pragma unroll
  for (int s = 0; s < 4; ++s) {
    #pragma unroll
    for (int r = 0; r < 8; ++r) {
      Out[(size_t)(m0 + hi * 8 + r) * D_MODEL + c0 + 16 * s + n] = acc[s][r];
    }
  }
}

extern "C" void kernel_launch(void* const* d_in, const int* in_sizes, int n_in,
                              void* d_out, int out_size, void* d_ws, size_t ws_size,
                              hipStream_t stream) {
  const float* q  = (const float*)d_in[0];
  const float* k  = (const float*)d_in[1];
  const float* v  = (const float*)d_in[2];
  const unsigned char* kpm = (const unsigned char*)d_in[3];
  const float* Wq = (const float*)d_in[4];
  const float* bq = (const float*)d_in[5];
  const float* Wk = (const float*)d_in[6];
  const float* bk = (const float*)d_in[7];
  const float* Wv = (const float*)d_in[8];
  const float* bv = (const float*)d_in[9];
  const float* Wo = (const float*)d_in[10];
  const float* bo = (const float*)d_in[11];
  float* out = (float*)d_out;

  const size_t headElems = (size_t)BATCH * N_HEADS * SEQ * D_HEAD;  // 4 Mi halves
  _Float16* qh  = (_Float16*)d_ws;
  _Float16* kh  = qh + headElems;
  _Float16* vh  = kh + headElems;
  _Float16* att = vh + headElems;   // total 32 MB of d_ws

  dim3 blk(256);  // 8 waves/block
  const int wavesProj = ((BATCH * SEQ) / 16) * (D_MODEL / 64);  // 4096
  dim3 grdProj(wavesProj / 8);

  qkv_proj_kernel<<<grdProj, blk, 0, stream>>>(q, Wq, bq, qh);
  qkv_proj_kernel<<<grdProj, blk, 0, stream>>>(k, Wk, bk, kh);
  qkv_proj_kernel<<<grdProj, blk, 0, stream>>>(v, Wv, bv, vh);

  // 32 (b,h) slices x 16 blocks, 8 q-tiles (waves) per block
  mha_attn_kernel<<<dim3(512), blk, 0, stream>>>(qh, kh, vh, kpm, att);

  out_proj_kernel<<<grdProj, blk, 0, stream>>>(att, Wo, bo, out);
}
